// GATConv_47974784697089
// MI455X (gfx1250) — matrix operand order repaired and checked by reading, not compiled, not run on hardware
//
#include <hip/hip_runtime.h>
#include <hip/hip_bf16.h>

typedef __attribute__((ext_vector_type(16))) _Float16 v16h;
typedef __attribute__((ext_vector_type(8)))  float    v8f;

#define N_NODES 50000
#define N_EDGES 800000
#define DIN     128
#define NHEADS  4
#define DHEAD   16
#define HD      64            // NHEADS*DHEAD
#define NEG_SLOPE 0.2f

// ---------- order-preserving float <-> uint key (for atomic segment max) ----
__device__ __forceinline__ unsigned fkey_enc(float f) {
  unsigned u = __float_as_uint(f);
  return (u & 0x80000000u) ? ~u : (u | 0x80000000u);
}
__device__ __forceinline__ float fkey_dec(unsigned k) {
  unsigned u = (k & 0x80000000u) ? (k & 0x7FFFFFFFu) : ~k;
  return __uint_as_float(u);
}

// ---------------------------------------------------------------------------
// Kernel 0: pre-swizzle W (f32, [DIN, 64]) into f16 WMMA B-fragments.
// B fragment layout (16-bit B 32x16, ISA 7.12.2): lane L -> col N = L%16,
//   hi = L/16; half e<8 holds K = 8*hi + e, half e>=8 holds K = 16 + 8*hi + (e-8).
// wfrag[((ks*4 + nt)*32 + lane)*16 + e]  (f16), 16 KB total.
// One thread per (ks, nt, lane) = 512 threads.
// ---------------------------------------------------------------------------
__global__ void gat_pack_w(const float* __restrict__ W,
                           _Float16* __restrict__ wfrag) {
  int idx = blockIdx.x * blockDim.x + threadIdx.x;
  if (idx >= 16 * 32) return;
  int lane = idx & 31;
  int ksnt = idx >> 5;          // 0..15
  int ks   = ksnt >> 2;
  int nt   = ksnt & 3;
  int m    = lane & 15;
  int hi   = lane >> 4;
  const float* wcol = W + (size_t)(ks * 32 + 8 * hi) * HD + nt * 16 + m;
  v16h b;
  #pragma unroll
  for (int e = 0; e < 16; ++e) b[e] = (_Float16)wcol[(e < 8 ? e : e + 8) * HD];
  *(v16h*)(wfrag + (size_t)idx * 16) = b;
}

// ---------------------------------------------------------------------------
// Kernel 1: feat = h @ W via v_wmma_f32_16x16x32_f16. One wave per 16-row tile.
// N_NODES = 3125 * 16 exactly, so no tail handling.
// A fragment layout (16-bit A 16x32): lane L -> row M = L%16, hi = L/16;
//   halves 0..7 hold K = 8*hi + i, halves 8..15 hold K = 16 + 8*hi + i
//   -> two contiguous 8-float runs per lane.
// C/D f32 layout: VGPR v -> row M = v + 8*hi, col N = L%16.
// B fragments come pre-swizzled (f16) from gat_pack_w.
// ---------------------------------------------------------------------------
__global__ __launch_bounds__(32)
void gat_gemm_wmma(const float* __restrict__ h,
                   const _Float16* __restrict__ wfrag,
                   float* __restrict__ feat) {
  const int row0 = blockIdx.x * 16;
  const int lane = threadIdx.x;
  const int m    = lane & 15;
  const int hi   = lane >> 4;

  v8f acc[4];
  #pragma unroll
  for (int nt = 0; nt < 4; ++nt) acc[nt] = (v8f){0.f,0.f,0.f,0.f,0.f,0.f,0.f,0.f};

  const float* arow  = h + (size_t)(row0 + m) * DIN;
  const v16h*  bfrag = (const v16h*)wfrag;   // indexed [(ks*4 + nt)*32 + lane]

  #pragma unroll
  for (int ks = 0; ks < 4; ++ks) {
    const int kbase = ks * 32;
    // ---- A fragment: two contiguous 8-float runs per lane, cvt to f16 ----
    v16h a;
    #pragma unroll
    for (int i = 0; i < 8; ++i) {
      a[i]     = (_Float16)arow[kbase + 8 * hi + i];
      a[8 + i] = (_Float16)arow[kbase + 16 + 8 * hi + i];
    }
    #pragma unroll
    for (int nt = 0; nt < 4; ++nt) {
      v16h b = bfrag[(ks * 4 + nt) * 32 + lane];
      acc[nt] = __builtin_amdgcn_wmma_f32_16x16x32_f16(
          /*neg_a=*/false, a, /*neg_b=*/false, b,
          /*c_mod=*/(short)0, acc[nt], /*reuse_a=*/false, /*reuse_b=*/false);
    }
  }

  // ---- store C tiles ----
  #pragma unroll
  for (int nt = 0; nt < 4; ++nt) {
    #pragma unroll
    for (int v = 0; v < 8; ++v) {
      feat[(size_t)(row0 + v + 8 * hi) * HD + nt * 16 + m] = acc[nt][v];
    }
  }
}

// ---------------------------------------------------------------------------
// Kernel 2: per-(node, head) attention halves el/er (16-wide dots).
// ---------------------------------------------------------------------------
__global__ void gat_el_er(const float* __restrict__ feat,
                          const float* __restrict__ attn_l,
                          const float* __restrict__ attn_r,
                          float* __restrict__ el,
                          float* __restrict__ er) {
  int idx = blockIdx.x * blockDim.x + threadIdx.x;
  if (idx >= N_NODES * NHEADS) return;
  int n  = idx >> 2;
  int hh = idx & 3;
  const float* f  = feat + (size_t)n * HD + hh * DHEAD;
  const float* al = attn_l + hh * DHEAD;
  const float* ar = attn_r + hh * DHEAD;
  float sl = 0.f, sr = 0.f;
  #pragma unroll
  for (int d = 0; d < DHEAD; ++d) {
    float fv = f[d];
    sl += fv * al[d];
    sr += fv * ar[d];
  }
  el[idx] = sl;
  er[idx] = sr;
}

// ---------------------------------------------------------------------------
// Kernel 3: zero/seed the reduction buffers (ws is poisoned by the harness).
// ---------------------------------------------------------------------------
__global__ void gat_init(unsigned* __restrict__ mkey,
                         float* __restrict__ denom,
                         float* __restrict__ accum) {
  int idx = blockIdx.x * blockDim.x + threadIdx.x;
  if (idx < N_NODES * NHEADS * DHEAD) accum[idx] = 0.f;
  if (idx < N_NODES * NHEADS) { mkey[idx] = 0u; denom[idx] = 0.f; }
}

// ---------------------------------------------------------------------------
// Kernel 4: per-(edge, head) leaky-relu logit + atomic segment max over dst.
// ---------------------------------------------------------------------------
__global__ void gat_edge_max(const int* __restrict__ src,
                             const int* __restrict__ dst,
                             const float* __restrict__ el,
                             const float* __restrict__ er,
                             float* __restrict__ ebuf,
                             unsigned* __restrict__ mkey) {
  int idx = blockIdx.x * blockDim.x + threadIdx.x;
  if (idx >= N_EDGES * NHEADS) return;
  int e  = idx >> 2;
  int hh = idx & 3;
  int s = src[e], d = dst[e];
  float ev = el[s * NHEADS + hh] + er[d * NHEADS + hh];
  ev = ev > 0.f ? ev : NEG_SLOPE * ev;
  ebuf[idx] = ev;
  atomicMax(&mkey[d * NHEADS + hh], fkey_enc(ev));
}

// ---------------------------------------------------------------------------
// Kernel 5: ex = exp(e - m[dst]); atomic segment sum into denom.
// ---------------------------------------------------------------------------
__global__ void gat_edge_exp(const int* __restrict__ dst,
                             const unsigned* __restrict__ mkey,
                             float* __restrict__ ebuf,
                             float* __restrict__ denom) {
  int idx = blockIdx.x * blockDim.x + threadIdx.x;
  if (idx >= N_EDGES * NHEADS) return;
  int e  = idx >> 2;
  int hh = idx & 3;
  int d = dst[e];
  float mv = fkey_dec(mkey[d * NHEADS + hh]);
  float ex = __expf(ebuf[idx] - mv);
  ebuf[idx] = ex;
  atomicAdd(&denom[d * NHEADS + hh], ex);
}

// ---------------------------------------------------------------------------
// Kernel 6: weighted aggregation: accum[dst,h,d] += (ex/denom) * feat[src,h,d].
// One thread per (edge, head, d) -> native global_atomic_add_f32, L2-resident.
// ---------------------------------------------------------------------------
__global__ void gat_aggregate(const int* __restrict__ src,
                              const int* __restrict__ dst,
                              const float* __restrict__ ebuf,
                              const float* __restrict__ denom,
                              const float* __restrict__ feat,
                              float* __restrict__ accum) {
  int idx = blockIdx.x * blockDim.x + threadIdx.x;
  if (idx >= N_EDGES * NHEADS * DHEAD) return;
  int d  = idx & 15;
  int hh = (idx >> 4) & 3;
  int e  = idx >> 6;
  int s = src[e], dd = dst[e];
  float a = ebuf[e * NHEADS + hh] / denom[dd * NHEADS + hh];
  float v = a * feat[(size_t)s * HD + hh * DHEAD + d];
  atomicAdd(&accum[(size_t)(dd * NHEADS + hh) * DHEAD + d], v);
}

// ---------------------------------------------------------------------------
// Kernel 7: mean over heads -> d_out [N, D].
// ---------------------------------------------------------------------------
__global__ void gat_finalize(const float* __restrict__ accum,
                             float* __restrict__ out) {
  int idx = blockIdx.x * blockDim.x + threadIdx.x;
  if (idx >= N_NODES * DHEAD) return;
  int n = idx >> 4;
  int d = idx & 15;
  const float* a = accum + (size_t)n * HD + d;
  out[idx] = 0.25f * (a[0] + a[DHEAD] + a[2 * DHEAD] + a[3 * DHEAD]);
}

extern "C" void kernel_launch(void* const* d_in, const int* in_sizes, int n_in,
                              void* d_out, int out_size, void* d_ws, size_t ws_size,
                              hipStream_t stream) {
  const float* h      = (const float*)d_in[0];   // [N, DIN]
  const float* W      = (const float*)d_in[1];   // [DIN, H*D]
  const float* attn_l = (const float*)d_in[2];   // [H, D]
  const float* attn_r = (const float*)d_in[3];   // [H, D]
  const int*   src    = (const int*)d_in[4];     // [E]
  const int*   dst    = (const int*)d_in[5];     // [E]
  float* out = (float*)d_out;                    // [N, D]

  // workspace layout (floats; every section is a multiple of 8 floats -> 32B aligned)
  float*     feat  = (float*)d_ws;                 // N*64     = 3,200,000
  float*     el    = feat  + (size_t)N_NODES * HD; // N*H      =   200,000
  float*     er    = el    + (size_t)N_NODES * NHEADS;
  unsigned*  mkey  = (unsigned*)(er + (size_t)N_NODES * NHEADS);
  float*     denom = (float*)(mkey + (size_t)N_NODES * NHEADS);
  float*     ebuf  = denom + (size_t)N_NODES * NHEADS;     // E*H   = 3,200,000
  float*     accum = ebuf  + (size_t)N_EDGES * NHEADS;     // N*H*D = 3,200,000
  _Float16*  wfrag = (_Float16*)(accum + (size_t)N_NODES * NHEADS * DHEAD); // 8192 halves

  // 0) pre-swizzle W into f16 WMMA B fragments (16 KB, block-invariant)
  gat_pack_w<<<2, 256, 0, stream>>>(W, wfrag);

  // 1) projection GEMM (WMMA), one wave per 16-row tile
  gat_gemm_wmma<<<N_NODES / 16, 32, 0, stream>>>(h, wfrag, feat);

  // 2) el / er per (node, head)
  {
    int t = N_NODES * NHEADS;
    gat_el_er<<<(t + 255) / 256, 256, 0, stream>>>(feat, attn_l, attn_r, el, er);
  }

  // 3) init reduction buffers
  {
    int t = N_NODES * NHEADS * DHEAD;
    gat_init<<<(t + 255) / 256, 256, 0, stream>>>(mkey, denom, accum);
  }

  // 4) edge logits + segment max
  {
    int t = N_EDGES * NHEADS;
    gat_edge_max<<<(t + 255) / 256, 256, 0, stream>>>(src, dst, el, er, ebuf, mkey);
  }

  // 5) exp + segment sum (denominator)
  {
    int t = N_EDGES * NHEADS;
    gat_edge_exp<<<(t + 255) / 256, 256, 0, stream>>>(dst, mkey, ebuf, denom);
  }

  // 6) weighted aggregation
  {
    int t = N_EDGES * NHEADS * DHEAD;
    gat_aggregate<<<(t + 255) / 256, 256, 0, stream>>>(src, dst, ebuf, denom, feat, accum);
  }

  // 7) head mean -> output
  {
    int t = N_NODES * DHEAD;
    gat_finalize<<<(t + 255) / 256, 256, 0, stream>>>(accum, out);
  }
}
